// LSTMNet_54881092108894
// MI455X (gfx1250) — compile-verified, matrix operand order checked by reading, not run on hardware
//
#include <hip/hip_runtime.h>

#define HID 32          // hidden dim
#define T_LEN 1024      // timesteps
#define B_TILE 16       // batch rows per wave (WMMA M)

typedef __attribute__((ext_vector_type(16))) _Float16 v16h;
typedef __attribute__((ext_vector_type(8)))  _Float16 v8h;
typedef __attribute__((ext_vector_type(2)))  _Float16 h2;
typedef __attribute__((ext_vector_type(8)))  float    v8f;

// Zero-instruction compiler fence: keeps the cross-lane LDS store->load
// transpose ordered in the instruction stream (hardware DS ops from one wave
// are in-order per ISA 7.3; the compiler inserts the precise dscnt waits).
#define LDS_FENCE() asm volatile("" ::: "memory")

// ---- activations: gfx1250 hardware tanh (v_tanh_f32) when available ----
#if __has_builtin(__builtin_amdgcn_tanhf)
__device__ __forceinline__ float tanh_hw(float x) { return __builtin_amdgcn_tanhf(x); }
#else
__device__ __forceinline__ float tanh_hw(float x) {
    return 2.0f * __builtin_amdgcn_rcpf(1.0f + __expf(-2.0f * x)) - 1.0f;
}
#endif
__device__ __forceinline__ float sigf(float x) {
    return 0.5f * tanh_hw(0.5f * x) + 0.5f;   // mul + v_tanh + fma
}

// K-storage permutation: storage slot q in [0,32) holds original hidden index
//   k(q) = (q>>1) + 16*(q&1)
// so hidden units col and col+16 are adjacent halfwords -> packed b32 h-stores.
// Legal because WMMA only needs A and B to agree on K ordering.
__device__ __forceinline__ int kOf(int q) { return (q >> 1) + ((q & 1) << 4); }

// Build one f16 WMMA B-operand tile (32x16) for gates = act @ W^T, K-permuted.
// 16-bit B layout (mirror of A): lane<16 holds q 0..7 / 16..23, lane>=16 holds
// q 8..15 / 24..31 (q0 = grp*8).
__device__ __forceinline__ v16h load_btile(const float* __restrict__ W, int j, int q0) {
    v16h b;
#pragma unroll
    for (int i = 0; i < 8; ++i) {
        b[i]     = (_Float16)W[j * HID + kOf(q0 + i)];
        b[i + 8] = (_Float16)W[j * HID + kOf(q0 + 16 + i)];
    }
    return b;
}

// Load one A-operand row (16x32 f16) from q-ordered f16 LDS (two ds_load_b128).
__device__ __forceinline__ v16h load_arow(const _Float16* lds, int row, int q0) {
    v8h lo = *(const v8h*)(lds + row * HID + q0);
    v8h hi = *(const v8h*)(lds + row * HID + q0 + 16);
    v16h a;
#pragma unroll
    for (int i = 0; i < 8; ++i) { a[i] = lo[i]; a[i + 8] = hi[i]; }
    return a;
}

__global__ __launch_bounds__(32, 1) void lstm2_persistent_kernel(
    const float* __restrict__ x,
    const float* __restrict__ Wih0, const float* __restrict__ Whh0,
    const float* __restrict__ bih0, const float* __restrict__ bhh0,
    const float* __restrict__ Wih1, const float* __restrict__ Whh1,
    const float* __restrict__ bih1, const float* __restrict__ bhh1,
    const float* __restrict__ Wlin, const float* __restrict__ blin,
    float* __restrict__ out, int T)
{
    __shared__ __align__(16) _Float16 lh0[B_TILE * HID];  // layer0 h (f16, q-ordered rows)
    __shared__ __align__(16) _Float16 lh1[B_TILE * HID];  // layer1 h

    const int lid = threadIdx.x & 31;
    const int col = lid & 15;       // tile column (N) / A row (M)
    const int grp = lid >> 4;       // lane half
    const int q0  = grp * 8;        // K-slot segment base for A/B operands
    const int mb  = grp * 8;        // M base for C/D layout (VGPR r -> M = mb + r)
    const int bbase = blockIdx.x * B_TILE;

    // ---- one-time weight staging: 33 f16 B-tiles resident in VGPRs ----
    v16h Bhh0[8], Bih1[8], Bhh1[8];
    v16h Bx[8];                     // rank-1 layer-0 input+bias: slot0=Wih0, slot1=b0
    v16h Bwl;                       // output head: Wlin in column N=0 only
    v8f  biasC1[8];                 // hoisted layer-1 C operand (bias splat)
#pragma unroll
    for (int n = 0; n < 8; ++n) {
        const int j = n * 16 + col;            // gate index this lane covers
        Bhh0[n] = load_btile(Whh0, j, q0);
        Bih1[n] = load_btile(Wih1, j, q0);
        Bhh1[n] = load_btile(Whh1, j, q0);
        v16h bx = {};
        if (grp == 0) {                        // K-slots 0,1 live in group-0 lanes
            bx[0] = (_Float16)Wih0[j];         // x coefficient
            bx[1] = (_Float16)(bih0[j] + bhh0[j]);  // bias via constant-1 slot
        }
        Bx[n] = bx;
        const float b1 = bih1[j] + bhh1[j];
#pragma unroll
        for (int r = 0; r < 8; ++r) biasC1[n][r] = b1;
    }
    {
        v16h bw = {};
        if (col == 0) {                        // only output column N=0 is real
#pragma unroll
            for (int i = 0; i < 8; ++i) {
                bw[i]     = (_Float16)Wlin[kOf(q0 + i)];
                bw[i + 8] = (_Float16)Wlin[kOf(q0 + 16 + i)];
            }
        }
        Bwl = bw;
    }
    v8f blC;
    { const float bl = blin[0];
#pragma unroll
      for (int r = 0; r < 8; ++r) blC[r] = bl; }
    const v8f zC = {};

    // A_x: constant part (1.0 at K-slot 1 for the bias row), x inserted per step
    v16h ax = {};
    if (grp == 0) ax[1] = (_Float16)1.0f;

    // zero h state in LDS, c state in VGPRs
    for (int i = lid; i < B_TILE * HID; i += 32) {
        lh0[i] = (_Float16)0.0f; lh1[i] = (_Float16)0.0f;
    }
    asm volatile("s_wait_dscnt 0x0" ::: "memory");

    v8f c0a = {}, c0b = {}, c1a = {}, c1b = {};

    const float* xrow = x + (size_t)(bbase + col) * T;
    float xv = xrow[0];                        // software-pipelined x

    for (int t = 0; t < T; ++t) {
        const float xnext = (t + 1 < T) ? xrow[t + 1] : 0.0f;

        // ---- step front: state loads + everything independent of this step's
        // ---- serial chain (layer-1 recurrent, previous output, x rank-1 term)
        v16h a0  = load_arow(lh0, col, q0);
        v16h a1h = load_arow(lh1, col, q0);     // h1 of step t-1

        // previous step's output head: one WMMA, column 0 = h1 . Wlin + bl
        v8f outD = __builtin_amdgcn_wmma_f32_16x16x32_f16(
            false, a1h, false, Bwl, (short)0, blC, false, false);

        v8f mid[8];
#pragma unroll
        for (int n = 0; n < 8; ++n) {
            mid[n] = __builtin_amdgcn_wmma_f32_16x16x32_f16(
                false, a1h, false, Bhh1[n], (short)0, biasC1[n], false, false);
        }

        // rank-1 x contribution + bias (x was loaded one step ahead)
        ax[0] = (grp == 0) ? (_Float16)xv : (_Float16)0.0f;
        v8f xacc[8];
#pragma unroll
        for (int n = 0; n < 8; ++n) {
            xacc[n] = __builtin_amdgcn_wmma_f32_16x16x32_f16(
                false, ax, false, Bx[n], (short)0, zC, false, false);
        }

        if (t != 0 && col == 0) {
#pragma unroll
            for (int r = 0; r < 8; ++r)
                out[(size_t)(bbase + mb + r) * T + (t - 1)] = outD[r];
        }

        // ================= layer 0: g = h0*Whh0^T + (x*Wih0^T + b) =================
        v8f acc[8];
#pragma unroll
        for (int n = 0; n < 8; ++n) {
            acc[n] = __builtin_amdgcn_wmma_f32_16x16x32_f16(
                false, a0, false, Bhh0[n], (short)0, xacc[n], false, false);
        }
        v8f h0a, h0b;
#pragma unroll
        for (int r = 0; r < 8; ++r) {
            float i0 = sigf(acc[0][r]), f0 = sigf(acc[2][r]);
            float g0 = tanh_hw(acc[4][r]), o0 = sigf(acc[6][r]);
            c0a[r] = f0 * c0a[r] + i0 * g0;  h0a[r] = o0 * tanh_hw(c0a[r]);
            float i1 = sigf(acc[1][r]), f1 = sigf(acc[3][r]);
            float g1 = tanh_hw(acc[5][r]), o1 = sigf(acc[7][r]);
            c0b[r] = f1 * c0b[r] + i1 * g1;  h0b[r] = o1 * tanh_hw(c0b[r]);
        }
        // h0 -> LDS: pack (col, col+16) pair into one b32 store per row
#pragma unroll
        for (int r = 0; r < 8; ++r) {
            h2 p; p[0] = (_Float16)h0a[r]; p[1] = (_Float16)h0b[r];
            *(h2*)(lh0 + (mb + r) * HID + 2 * col) = p;
        }
        LDS_FENCE();   // DS in-order: transposed re-load below sees the stores

        // ============ layer 1: g = h0*Wih1^T + mid (chained WMMA) ============
        v16h a1i = load_arow(lh0, col, q0);
#pragma unroll
        for (int n = 0; n < 8; ++n) {
            acc[n] = __builtin_amdgcn_wmma_f32_16x16x32_f16(
                false, a1i, false, Bih1[n], (short)0, mid[n], false, false);
        }
        v8f h1a, h1b;
#pragma unroll
        for (int r = 0; r < 8; ++r) {
            float i0 = sigf(acc[0][r]), f0 = sigf(acc[2][r]);
            float g0 = tanh_hw(acc[4][r]), o0 = sigf(acc[6][r]);
            c1a[r] = f0 * c1a[r] + i0 * g0;  h1a[r] = o0 * tanh_hw(c1a[r]);
            float i1 = sigf(acc[1][r]), f1 = sigf(acc[3][r]);
            float g1 = tanh_hw(acc[5][r]), o1 = sigf(acc[7][r]);
            c1b[r] = f1 * c1b[r] + i1 * g1;  h1b[r] = o1 * tanh_hw(c1b[r]);
        }

        // h1 -> LDS (packed b32) for next step's a1h / output head
#pragma unroll
        for (int r = 0; r < 8; ++r) {
            h2 p; p[0] = (_Float16)h1a[r]; p[1] = (_Float16)h1b[r];
            *(h2*)(lh1 + (mb + r) * HID + 2 * col) = p;
        }
        LDS_FENCE();

        xv = xnext;
    }

    // ---- epilogue: output for the final step (t = T-1) ----
    {
        v16h af = load_arow(lh1, col, q0);
        v8f outD = __builtin_amdgcn_wmma_f32_16x16x32_f16(
            false, af, false, Bwl, (short)0, blC, false, false);
        if (col == 0) {
#pragma unroll
            for (int r = 0; r < 8; ++r)
                out[(size_t)(bbase + mb + r) * T + (T - 1)] = outD[r];
        }
    }
}

extern "C" void kernel_launch(void* const* d_in, const int* in_sizes, int n_in,
                              void* d_out, int out_size, void* d_ws, size_t ws_size,
                              hipStream_t stream) {
    (void)n_in; (void)d_ws; (void)ws_size; (void)out_size;
    const float* x    = (const float*)d_in[0];
    const float* Wih0 = (const float*)d_in[1];
    const float* Whh0 = (const float*)d_in[2];
    const float* bih0 = (const float*)d_in[3];
    const float* bhh0 = (const float*)d_in[4];
    const float* Wih1 = (const float*)d_in[5];
    const float* Whh1 = (const float*)d_in[6];
    const float* bih1 = (const float*)d_in[7];
    const float* bhh1 = (const float*)d_in[8];
    const float* Wlin = (const float*)d_in[9];
    const float* blin = (const float*)d_in[10];
    float* out = (float*)d_out;

    const int T = T_LEN;
    const int B = in_sizes[0] / T;          // 2048
    const int nblocks = B / B_TILE;         // 128 waves, each owns 16 batch rows

    lstm2_persistent_kernel<<<dim3(nblocks), dim3(32), 0, stream>>>(
        x, Wih0, Whh0, bih0, bhh0, Wih1, Whh1, bih1, bhh1, Wlin, blin, out, T);
}
